// Mamba2_3504693314042
// MI455X (gfx1250) — matrix operand order repaired and checked by reading, not compiled, not run on hardware
//
#include <hip/hip_runtime.h>
#include <hip/hip_bf16.h>
#include <cstdint>

#define DMODEL  2048
#define DINNER  4096
#define DXB     1024
#define DSTATE  128
#define HDIM    64
#define NHEADS  64
#define CONVDIM 6144
#define DPROJ   10304
#define CHUNKL  256
#define NCHUNK  8
#define SEQLEN  2048
#define BATCH   2
#define NTOK    4096
#define SMEM_INTRA (65536 + 32768 + 16384 + 3072)

typedef __attribute__((ext_vector_type(16))) _Float16 v16h;
typedef __attribute__((ext_vector_type(4)))  _Float16 v4h;
typedef __attribute__((ext_vector_type(8)))  float    v8f;

static __device__ __forceinline__ v8f wmma16(v16h a, v16h b, v8f c) {
  // v_wmma_f32_16x16x32_f16: (neg_a, A, neg_b, B, c_mod, C, reuse_a, reuse_b)
  return __builtin_amdgcn_wmma_f32_16x16x32_f16(false, a, false, b, (short)0, c, false, false);
}
static __device__ __forceinline__ float siluf(float x) { return x / (1.f + __expf(-x)); }
static __device__ __forceinline__ void pack8(v16h& f, int base, float4 a, float4 b) {
  f[base+0]=(_Float16)a.x; f[base+1]=(_Float16)a.y; f[base+2]=(_Float16)a.z; f[base+3]=(_Float16)a.w;
  f[base+4]=(_Float16)b.x; f[base+5]=(_Float16)b.y; f[base+6]=(_Float16)b.z; f[base+7]=(_Float16)b.w;
}
// Async DMA: 16 bytes global -> LDS, tracked by ASYNCcnt (no VGPR data path).
static __device__ __forceinline__ void async_ld16(unsigned lds_off, const float* gptr) {
  asm volatile("global_load_async_to_lds_b128 %0, %1, off"
               :: "v"(lds_off), "v"((unsigned long long)(uintptr_t)gptr)
               : "memory");
}

// ---------------------------------------------------------------------------
// Generic GEMM: C[M,N] = A[M,K] * B[N,K]^T, fp32 memory, f16 WMMA math.
// Block = 256 thr (8 waves), tile 128(M) x 64(N), K-step 32.
// GLOBAL_LOAD_ASYNC_TO_LDS double-buffers raw f32 tiles; one conversion pass
// produces f16 tiles so WMMA fragments are clean 2x ds_load_b128 reads.
// ---------------------------------------------------------------------------
__global__ __launch_bounds__(256) void gemm_tn(const float* __restrict__ A,
                                               const float* __restrict__ B,
                                               float* __restrict__ C,
                                               int M, int N, int K) {
  __shared__ __align__(16) float    stA[2][128 * 32];   // 32KB  DMA staging (f32)
  __shared__ __align__(16) float    stB[2][64 * 32];    // 16KB
  __shared__ __align__(16) _Float16 sA16[128 * 32];     // 8KB   WMMA tiles (f16)
  __shared__ __align__(16) _Float16 sB16[64 * 32];      // 4KB
  const int m0 = blockIdx.x * 128, n0 = blockIdx.y * 64;
  const int tid = threadIdx.x, lane = tid & 31, wave = tid >> 5;
  const int khalf = (lane >> 4) * 8, kb16 = (lane >> 4) * 16;
  const unsigned baseA[2] = { (unsigned)(uintptr_t)&stA[0][0], (unsigned)(uintptr_t)&stA[1][0] };
  const unsigned baseB[2] = { (unsigned)(uintptr_t)&stB[0][0], (unsigned)(uintptr_t)&stB[1][0] };

  auto issue = [&](int buf, int k0) {
#pragma unroll
    for (int it = 0; it < 4; ++it) {            // A tile: 128x32 f32 = 16KB
      int e = (it * 256 + tid) * 4;             // element index, 4 floats per op
      int r = e >> 5, cc2 = e & 31;
      async_ld16(baseA[buf] + (unsigned)e * 4u, A + (size_t)(m0 + r) * K + k0 + cc2);
    }
#pragma unroll
    for (int it = 0; it < 2; ++it) {            // B tile: 64x32 f32 = 8KB
      int e = (it * 256 + tid) * 4;
      int r = e >> 5, cc2 = e & 31;
      async_ld16(baseB[buf] + (unsigned)e * 4u, B + (size_t)(n0 + r) * K + k0 + cc2);
    }
  };

  v8f acc[4] = {};
  issue(0, 0);
  const int nIter = K / 32;
  for (int i = 0; i < nIter; ++i) {
    if (i + 1 < nIter) {
      issue((i + 1) & 1, (i + 1) * 32);         // next tile streams into other buffer
      // 6 newest (next-buffer) ops may stay in flight; older batch must be done.
      asm volatile("s_wait_asynccnt 0x6" ::: "memory");
    } else {
      asm volatile("s_wait_asynccnt 0x0" ::: "memory");
    }
    __syncthreads();                            // all waves' DMA for this buffer landed
    // one-shot f32 -> f16 conversion pass (each element converted exactly once)
    const float* fA = stA[i & 1];
    const float* fB = stB[i & 1];
#pragma unroll
    for (int it = 0; it < 4; ++it) {
      int e = (it * 256 + tid) * 4;
      float4 v = *(const float4*)(fA + e);
      v4h hv = { (_Float16)v.x, (_Float16)v.y, (_Float16)v.z, (_Float16)v.w };
      *(v4h*)(sA16 + e) = hv;
    }
#pragma unroll
    for (int it = 0; it < 2; ++it) {
      int e = (it * 256 + tid) * 4;
      float4 v = *(const float4*)(fB + e);
      v4h hv = { (_Float16)v.x, (_Float16)v.y, (_Float16)v.z, (_Float16)v.w };
      *(v4h*)(sB16 + e) = hv;
    }
    __syncthreads();
    v16h a;
    const int mr = wave * 16 + (lane & 15);
#pragma unroll
    for (int cc = 0; cc < 16; ++cc)
      a[cc] = sA16[mr * 32 + kb16 + cc];
#pragma unroll
    for (int nt = 0; nt < 4; ++nt) {
      v16h bf;
      const int nr = nt * 16 + (lane & 15);
#pragma unroll
      for (int cc = 0; cc < 16; ++cc) bf[cc] = sB16[nr * 32 + kb16 + cc];
      acc[nt] = wmma16(a, bf, acc[nt]);
    }
    __syncthreads();                            // f16 tiles + staging free for reuse
  }
  const int mrow = m0 + wave * 16 + ((lane >> 4) ? 8 : 0);
  const int nc0 = n0 + (lane & 15);
#pragma unroll
  for (int nt = 0; nt < 4; ++nt)
#pragma unroll
    for (int vi = 0; vi < 8; ++vi)
      C[(size_t)(mrow + vi) * N + nc0 + nt * 16] = acc[nt][vi];
}

// ---------------------------------------------------------------------------
// Depthwise causal conv1d (k=4) + SiLU over xBC channels of zxbcdt.
// ---------------------------------------------------------------------------
__global__ void conv_silu_k(const float* __restrict__ zx, const float* __restrict__ cw,
                            const float* __restrict__ cb, float* __restrict__ xbc) {
  size_t idx = (size_t)blockIdx.x * 256 + threadIdx.x;
  int ch = (int)(idx % CONVDIM);
  size_t tok = idx / CONVDIM;
  int l = (int)(tok % SEQLEN);
  float acc = cb[ch];
#pragma unroll
  for (int k = 0; k < 4; ++k) {
    int ls = l - 3 + k;
    if (ls >= 0) acc += zx[(tok - 3 + k) * (size_t)DPROJ + DINNER + ch] * cw[ch * 4 + k];
  }
  xbc[idx] = siluf(acc);
}

__global__ void dt_softplus_k(const float* __restrict__ zx, const float* __restrict__ dt_bias,
                              float* __restrict__ dtsp) {
  size_t idx = (size_t)blockIdx.x * 256 + threadIdx.x;
  int h = (int)(idx & 63);
  size_t tok = idx >> 6;
  float x = zx[tok * (size_t)DPROJ + (DINNER + CONVDIM) + h] + dt_bias[h];
  dtsp[idx] = (x > 20.f) ? x : log1pf(__expf(x));
}

// ---------------------------------------------------------------------------
// Intra-chunk SSD: per (chunk c, head h, batch b). Computes dA cumsum, Y_diag
// (+ D skip) and per-chunk states via f16 WMMA. Dynamic LDS ~115KB.
// ---------------------------------------------------------------------------
__global__ __launch_bounds__(256) void scan_intra(const float* __restrict__ xbc,
                                                  const float* __restrict__ dtsp,
                                                  const float* __restrict__ A_log,
                                                  const float* __restrict__ Dvec,
                                                  float* __restrict__ dacs_g,
                                                  float* __restrict__ states,
                                                  float* __restrict__ ybuf) {
  extern __shared__ char smem[];
  _Float16 (*sBc)[DSTATE] = (_Float16(*)[DSTATE])smem;                       // 64KB [s][n]
  _Float16 (*sXdt)[CHUNKL] = (_Float16(*)[CHUNKL])(smem + 65536);            // 32KB [p][s]
  _Float16* glBase = (_Float16*)(smem + 98304);                              // 16KB (8 waves x 32x32)
  float* sDacs  = (float*)(smem + 98304 + 16384);
  float* sDecay = sDacs + CHUNKL;
  float* sDt    = sDecay + CHUNKL;

  const int c = blockIdx.x, h = blockIdx.y, b = blockIdx.z;
  const int tid = threadIdx.x, lane = tid & 31, wave = tid >> 5;
  const int khalf = (lane >> 4) * 8, kb16 = (lane >> 4) * 16;
  const size_t tokBase = (size_t)b * SEQLEN + (size_t)c * CHUNKL;
  const float Ah = -__expf(A_log[h]);
  const int chC = 2 * DXB + (h >> 1) * DSTATE;

  // prefetch the C rows consumed in phase 2 (one row segment per thread)
  __builtin_prefetch(xbc + (tokBase + tid) * (size_t)CONVDIM + chC, 0, 3);

  // ---- phase 0: dA cumsum ----
  {
    float dtv = dtsp[(tokBase + tid) * NHEADS + h];
    sDt[tid] = dtv;
    sDacs[tid] = dtv * Ah;
  }
  __syncthreads();
  for (int off = 1; off < CHUNKL; off <<= 1) {
    float v = sDacs[tid];
    float u = (tid >= off) ? sDacs[tid - off] : 0.f;
    __syncthreads();
    sDacs[tid] = v + u;
    __syncthreads();
  }
  dacs_g[(((size_t)b * NHEADS + h) * NCHUNK + c) * CHUNKL + tid] = sDacs[tid];
  sDecay[tid] = __expf(sDacs[CHUNKL - 1] - sDacs[tid]);

  // ---- phase 1: stage Bc[s][n] and xdt^T[p][s] in LDS as f16 ----
  const int chB = DXB + (h >> 3) * DSTATE;
  for (int idx = tid; idx < CHUNKL * DSTATE; idx += 256) {
    int s = idx >> 7, n = idx & 127;
    sBc[s][n] = (_Float16)xbc[(tokBase + s) * (size_t)CONVDIM + chB + n];
  }
  const int chX0 = (h >> 3) * 128 + (h & 1) * 64;
  for (int idx = tid; idx < HDIM * CHUNKL; idx += 256) {
    int p = idx >> 8, s = idx & 255;
    sXdt[p][s] = (_Float16)(xbc[(tokBase + s) * (size_t)CONVDIM + chX0 + p] * sDt[s]);
  }
  __syncthreads();

  // ---- phase 2: per-wave 32-row t-strip: G = C*B^T, Y_diag = (G.L)*xdt ----
  const int t0 = wave * 32;
  v16h caf[2][4];
#pragma unroll
  for (int tr = 0; tr < 2; ++tr) {
    const float* rp = xbc + (tokBase + t0 + tr * 16 + (lane & 15)) * (size_t)CONVDIM + chC;
#pragma unroll
    for (int kb = 0; kb < 4; ++kb) {
      float4 a0 = *(const float4*)(rp + kb * 32 + khalf);
      float4 a1 = *(const float4*)(rp + kb * 32 + khalf + 4);
      float4 b0 = *(const float4*)(rp + kb * 32 + 16 + khalf);
      float4 b1 = *(const float4*)(rp + kb * 32 + 16 + khalf + 4);
      v16h f; pack8(f, 0, a0, a1); pack8(f, 8, b0, b1);
      caf[tr][kb] = f;
    }
  }
  _Float16 (*myGL)[32] = (_Float16(*)[32])(glBase + wave * 1024);
  v8f yacc[2][4] = {};
  for (int sb = 0; sb < NCHUNK; ++sb) {
    const int s0 = sb * 32;
#pragma unroll
    for (int tr = 0; tr < 2; ++tr) {
#pragma unroll
      for (int sc = 0; sc < 2; ++sc) {
        v8f g = {};
        const int scol = s0 + sc * 16 + (lane & 15);
#pragma unroll
        for (int kb = 0; kb < 4; ++kb) {
          v16h bf;
#pragma unroll
          for (int cc = 0; cc < 16; ++cc) bf[cc] = sBc[scol][kb * 32 + (cc >> 3) * 16 + khalf + (cc & 7)];
          g = wmma16(caf[tr][kb], bf, g);
        }
        const int trow = t0 + tr * 16 + ((lane >> 4) ? 8 : 0);
        const float dcs = sDacs[scol];
#pragma unroll
        for (int vi = 0; vi < 8; ++vi) {
          int t = trow + vi;
          float lv = (scol <= t) ? __expf(sDacs[t] - dcs) : 0.f;
          myGL[tr * 16 + ((lane >> 4) ? 8 : 0) + vi][sc * 16 + (lane & 15)] = (_Float16)(g[vi] * lv);
        }
      }
    }
    asm volatile("s_wait_dscnt 0x0" ::: "memory");  // wave-local LDS RAW fence
#pragma unroll
    for (int tr = 0; tr < 2; ++tr) {
      v16h af;
      const int r = tr * 16 + (lane & 15);
#pragma unroll
      for (int cc = 0; cc < 16; ++cc) af[cc] = myGL[r][kb16 + cc];
#pragma unroll
      for (int pc = 0; pc < 4; ++pc) {
        v16h bf;
        const int pcol = pc * 16 + (lane & 15);
#pragma unroll
        for (int cc = 0; cc < 16; ++cc) bf[cc] = sXdt[pcol][s0 + kb16 + cc];
        yacc[tr][pc] = wmma16(af, bf, yacc[tr][pc]);
      }
    }
  }
  const float Dh = Dvec[h];
#pragma unroll
  for (int tr = 0; tr < 2; ++tr) {
    const int trow = t0 + tr * 16 + ((lane >> 4) ? 8 : 0);
#pragma unroll
    for (int pc = 0; pc < 4; ++pc) {
      const int p = pc * 16 + (lane & 15);
      const size_t colg = (size_t)h * HDIM + p;
#pragma unroll
      for (int vi = 0; vi < 8; ++vi) {
        const size_t tok = tokBase + trow + vi;
        float xv = xbc[tok * (size_t)CONVDIM + chX0 + p];
        ybuf[tok * (size_t)DINNER + colg] = yacc[tr][pc][vi] + xv * Dh;
      }
    }
  }

  // ---- phase 3: states[p,n] = sum_s decay[s]*xdt[s,p]*Bc[s,n] ----
  const int pst = (wave & 3) * 16;
  const int nh0 = (wave >> 2) * 64;
  v8f sacc[4] = {};
  for (int kb = 0; kb < 8; ++kb) {
    v16h af;
    const int prow = pst + (lane & 15);
#pragma unroll
    for (int cc = 0; cc < 16; ++cc) {
      int s = kb * 32 + kb16 + cc;
      af[cc] = (_Float16)((float)sXdt[prow][s] * sDecay[s]);
    }
#pragma unroll
    for (int nt = 0; nt < 4; ++nt) {
      v16h bf;
      const int ncol = nh0 + nt * 16 + (lane & 15);
#pragma unroll
      for (int cc = 0; cc < 16; ++cc) bf[cc] = sBc[kb * 32 + kb16 + cc][ncol];
      sacc[nt] = wmma16(af, bf, sacc[nt]);
    }
  }
  const size_t sb0 = (((size_t)b * NCHUNK + c) * NHEADS + h) * (HDIM * DSTATE);
  const int prow2 = pst + ((lane >> 4) ? 8 : 0);
#pragma unroll
  for (int nt = 0; nt < 4; ++nt) {
    const int n = nh0 + nt * 16 + (lane & 15);
#pragma unroll
    for (int vi = 0; vi < 8; ++vi)
      states[sb0 + (size_t)(prow2 + vi) * DSTATE + n] = sacc[nt][vi];
  }
}

// ---------------------------------------------------------------------------
// Inter-chunk recurrence: prev[c] = carry; carry = carry*exp(dAcs_last[c]) + states[c]
// ---------------------------------------------------------------------------
__global__ __launch_bounds__(256) void scan_inter(const float* __restrict__ states,
                                                  const float* __restrict__ dacs_g,
                                                  float* __restrict__ prevs) {
  const int h = blockIdx.x, b = blockIdx.y;
  const int tid = threadIdx.x;
  float carry[32];
#pragma unroll
  for (int j = 0; j < 32; ++j) carry[j] = 0.f;
  for (int c = 0; c < NCHUNK; ++c) {
    const size_t base = (((size_t)b * NCHUNK + c) * NHEADS + h) * (HDIM * DSTATE);
    const float cdec = __expf(dacs_g[(((size_t)b * NHEADS + h) * NCHUNK + c) * CHUNKL + (CHUNKL - 1)]);
#pragma unroll
    for (int j = 0; j < 32; ++j) {
      const size_t e = (size_t)j * 256 + tid;
      prevs[base + e] = carry[j];
      carry[j] = carry[j] * cdec + states[base + e];
    }
  }
}

// ---------------------------------------------------------------------------
// Off-diagonal: Y_off[t,p] = exp(dAcs[t]) * sum_n C[t,n]*prev[p,n], added to y.
// ---------------------------------------------------------------------------
__global__ __launch_bounds__(256) void scan_off(const float* __restrict__ xbc,
                                                const float* __restrict__ dacs_g,
                                                const float* __restrict__ prevs,
                                                float* __restrict__ ybuf) {
  const int c = blockIdx.x, h = blockIdx.y, b = blockIdx.z;
  const int tid = threadIdx.x, lane = tid & 31, wave = tid >> 5;
  const int khalf = (lane >> 4) * 8;
  const size_t tokBase = (size_t)b * SEQLEN + (size_t)c * CHUNKL;
  const int chC = 2 * DXB + (h >> 1) * DSTATE;
  const int t0 = wave * 32;
  v16h caf[2][4];
#pragma unroll
  for (int tr = 0; tr < 2; ++tr) {
    const float* rp = xbc + (tokBase + t0 + tr * 16 + (lane & 15)) * (size_t)CONVDIM + chC;
#pragma unroll
    for (int kb = 0; kb < 4; ++kb) {
      float4 a0 = *(const float4*)(rp + kb * 32 + 2 * khalf);
      float4 a1 = *(const float4*)(rp + kb * 32 + 2 * khalf + 4);
      float4 b0 = *(const float4*)(rp + kb * 32 + 2 * khalf + 8);
      float4 b1 = *(const float4*)(rp + kb * 32 + 2 * khalf + 12);
      v16h f; pack8(f, 0, a0, a1); pack8(f, 8, b0, b1);
      caf[tr][kb] = f;
    }
  }
  const float* pr = prevs + (((size_t)b * NCHUNK + c) * NHEADS + h) * (HDIM * DSTATE);
  v8f acc[2][4] = {};
#pragma unroll
  for (int pc = 0; pc < 4; ++pc) {
    const int pcol = pc * 16 + (lane & 15);
#pragma unroll
    for (int kb = 0; kb < 4; ++kb) {
      const float* bp = pr + (size_t)pcol * DSTATE + kb * 32 + (lane >> 4) * 16;
      float4 q0 = ((const float4*)bp)[0], q1 = ((const float4*)bp)[1];
      float4 q2 = ((const float4*)bp)[2], q3 = ((const float4*)bp)[3];
      v16h bf; pack8(bf, 0, q0, q1); pack8(bf, 8, q2, q3);
#pragma unroll
      for (int tr = 0; tr < 2; ++tr) acc[tr][pc] = wmma16(caf[tr][kb], bf, acc[tr][pc]);
    }
  }
  const float* dca = dacs_g + (((size_t)b * NHEADS + h) * NCHUNK + c) * CHUNKL;
#pragma unroll
  for (int tr = 0; tr < 2; ++tr) {
    const int trow = t0 + tr * 16 + ((lane >> 4) ? 8 : 0);
#pragma unroll
    for (int pc = 0; pc < 4; ++pc) {
      const int p = pc * 16 + (lane & 15);
      const size_t colg = (size_t)h * HDIM + p;
#pragma unroll
      for (int vi = 0; vi < 8; ++vi) {
        const int t = trow + vi;
        const size_t o = (tokBase + t) * (size_t)DINNER + colg;
        ybuf[o] += acc[tr][pc][vi] * __expf(dca[t]);
      }
    }
  }
}

// ---------------------------------------------------------------------------
// Gated grouped RMSNorm (group 128), in place on ybuf.
// ---------------------------------------------------------------------------
__global__ __launch_bounds__(128) void gate_norm(float* __restrict__ ybuf,
                                                 const float* __restrict__ zx,
                                                 const float* __restrict__ w) {
  const size_t tok = (size_t)blockIdx.x >> 5;
  const int grp = blockIdx.x & 31;
  const int tid = threadIdx.x;
  const size_t e = (size_t)grp * 128 + tid;
  const float zv = zx[tok * (size_t)DPROJ + e];
  const float v = ybuf[tok * (size_t)DINNER + e] * siluf(zv);
  float ss = v * v;
#pragma unroll
  for (int off = 16; off; off >>= 1) ss += __shfl_xor(ss, off, 32);
  __shared__ float red[4];
  if ((tid & 31) == 0) red[tid >> 5] = ss;
  __syncthreads();
  const float tot = red[0] + red[1] + red[2] + red[3];
  const float rstd = rsqrtf(tot / 128.f + 1e-5f);
  ybuf[tok * (size_t)DINNER + e] = v * rstd * w[e];
}

// ---------------------------------------------------------------------------
extern "C" void kernel_launch(void* const* d_in, const int* in_sizes, int n_in,
                              void* d_out, int out_size, void* d_ws, size_t ws_size,
                              hipStream_t stream) {
  (void)in_sizes; (void)n_in; (void)out_size; (void)ws_size;
  const float* u       = (const float*)d_in[0];
  const float* W_in    = (const float*)d_in[1];
  const float* conv_w  = (const float*)d_in[2];
  const float* conv_b  = (const float*)d_in[3];
  const float* dt_bias = (const float*)d_in[4];
  const float* A_log   = (const float*)d_in[5];
  const float* Dv      = (const float*)d_in[6];
  const float* norm_w  = (const float*)d_in[7];
  const float* W_out   = (const float*)d_in[8];
  float* out = (float*)d_out;

  float* zx     = (float*)d_ws;                                            // 4096 x 10304
  float* xbc    = zx     + (size_t)NTOK * DPROJ;                           // 4096 x 6144
  float* dtsp   = xbc    + (size_t)NTOK * CONVDIM;                         // 4096 x 64
  float* dacs   = dtsp   + (size_t)NTOK * NHEADS;                          // b,h,c,256
  float* states = dacs   + (size_t)BATCH * NHEADS * NCHUNK * CHUNKL;       // b,c,h,64,128
  float* prevs  = states + (size_t)BATCH * NCHUNK * NHEADS * HDIM * DSTATE;
  float* ybuf   = prevs  + (size_t)BATCH * NCHUNK * NHEADS * HDIM * DSTATE; // 4096 x 4096

  (void)hipFuncSetAttribute((const void*)scan_intra,
                            hipFuncAttributeMaxDynamicSharedMemorySize, SMEM_INTRA);

  gemm_tn<<<dim3(NTOK / 128, DPROJ / 64), 256, 0, stream>>>(u, W_in, zx, NTOK, DPROJ, DMODEL);
  conv_silu_k<<<(NTOK * (size_t)CONVDIM) / 256, 256, 0, stream>>>(zx, conv_w, conv_b, xbc);
  dt_softplus_k<<<(NTOK * NHEADS) / 256, 256, 0, stream>>>(zx, dt_bias, dtsp);
  scan_intra<<<dim3(NCHUNK, NHEADS, BATCH), 256, SMEM_INTRA, stream>>>(xbc, dtsp, A_log, Dv,
                                                                       dacs, states, ybuf);
  scan_inter<<<dim3(NHEADS, BATCH), 256, 0, stream>>>(states, dacs, prevs);
  scan_off<<<dim3(NCHUNK, NHEADS, BATCH), 256, 0, stream>>>(xbc, dacs, prevs, ybuf);
  gate_norm<<<NTOK * 32, 128, 0, stream>>>(ybuf, zx, norm_w);
  gemm_tn<<<dim3(NTOK / 128, DMODEL / 64), 256, 0, stream>>>(ybuf, W_out, out, NTOK, DMODEL, DINNER);
}
